// selflstm_22084721836639
// MI455X (gfx1250) — matrix operand order repaired and checked by reading, not compiled, not run on hardware
//
#include <hip/hip_runtime.h>
#include <hip/hip_fp16.h>

typedef __attribute__((ext_vector_type(16))) _Float16 v16h;
typedef __attribute__((ext_vector_type(8)))  _Float16 v8h;
typedef __attribute__((ext_vector_type(8)))  float    v8f;

#define CDIM 1024
#define LDIM 2048
#define BDIM 8
#define GDIM 4096          // 4*C
#define NWG  64            // workgroups in the scan (16 channels each)

// ---- workspace layout (bytes, all 256-aligned) ----
#define WS_X16   ((size_t)0)                       // 16384*1024 fp16 = 32 MB
#define WS_WIH   ((size_t)33554432)                // 4096*1024 fp16 packed = 8 MB
#define WS_WHH   ((size_t)41943040)                // 64*4*32*32*16 fp16 packed = 8 MB
#define WS_BIAS  ((size_t)50331648)                // 4096 fp32
#define WS_HBUF  ((size_t)50348032)                // 2*16*1024 fp16 (double-buffered h)
#define WS_BAR   ((size_t)50413568)                // 2048 int barrier counters
#define WS_XPROJ ((size_t)50421760)                // 2048*8*4096 fp32 = 256 MB

static __device__ inline v16h combine16(v8h lo, v8h hi) {
    v16h r;
#pragma unroll
    for (int i = 0; i < 8; ++i) { r[i] = lo[i]; r[i + 8] = hi[i]; }
    return r;
}

// ---------------- prep kernels ----------------
__global__ void k_cvt_x(const float* __restrict__ x, _Float16* __restrict__ x16, int n) {
    int i = blockIdx.x * blockDim.x + threadIdx.x;
    if (i < n) x16[i] = (_Float16)x[i];
}

// Pack W_ih into B-fragment order: frag[(nt*32+kt)*32 + lane][e] = W[nt*16+(lane&15)][kt*32+8*(lane>>4)+(e&7)+16*(e>>3)]
__global__ void k_pack_wih(const float* __restrict__ W, _Float16* __restrict__ F) {
    int i = blockIdx.x * blockDim.x + threadIdx.x;
    if (i >= GDIM * CDIM) return;
    int e = i & 15, lane = (i >> 4) & 31, kt = (i >> 9) & 31, nt = i >> 14; // nt 0..255
    int n = nt * 16 + (lane & 15), k = kt * 32 + (lane >> 4) * 8 + (e & 7) + (e >> 3) * 16;
    F[i] = (_Float16)W[(size_t)n * CDIM + k];
}

// Pack W_hh per-workgroup slice: row = gate*1024 + w*16 + (lane&15)
__global__ void k_pack_whh(const float* __restrict__ W, _Float16* __restrict__ F) {
    int i = blockIdx.x * blockDim.x + threadIdx.x;
    if (i >= GDIM * CDIM) return;
    int e = i & 15, lane = (i >> 4) & 31, kt = (i >> 9) & 31, nt = (i >> 14) & 3, w = i >> 16;
    int r = nt * CDIM + w * 16 + (lane & 15), k = kt * 32 + (lane >> 4) * 8 + (e & 7) + (e >> 3) * 16;
    F[i] = (_Float16)W[(size_t)r * CDIM + k];
}

__global__ void k_misc(const float* __restrict__ bih, const float* __restrict__ bhh,
                       float* __restrict__ bias, _Float16* __restrict__ hbuf,
                       int* __restrict__ bar) {
    int i = blockIdx.x * blockDim.x + threadIdx.x;
    if (i < GDIM) bias[i] = bih[i] + bhh[i];
    if (i < 2 * 16 * CDIM) hbuf[i] = (_Float16)0.f;
    if (i < LDIM) bar[i] = 0;
}

// ---------------- x-projection GEMM: x_proj[t][b][g] = x @ W_ih^T + bias ----------------
// 4-way N-tile register blocking: 1 A-fragment load feeds 4 WMMAs.
__global__ __launch_bounds__(256) void k_xproj(const _Float16* __restrict__ x16,
                                               const _Float16* __restrict__ wf,
                                               const float* __restrict__ bias,
                                               float* __restrict__ xp) {
    const int lane = threadIdx.x & 31;
    const int wv   = threadIdx.x >> 5;
    const int mt   = blockIdx.x;                   // 0..1023 (M tiles; m = b*2048+l)
    const int kb   = (lane >> 4) * 8;
    const _Float16* arow = x16 + (size_t)(mt * 16 + (lane & 15)) * CDIM + kb;

    for (int j = 0; j < 8; ++j) {
        const int nt0 = (j * 8 + wv) * 4;          // 4 consecutive N tiles
        v8f acc0 = {}, acc1 = {}, acc2 = {}, acc3 = {};
#pragma unroll 2
        for (int kt = 0; kt < 32; ++kt) {
            v16h a = combine16(*(const v8h*)(arow + kt * 32),
                               *(const v8h*)(arow + kt * 32 + 16));
            const _Float16* bp = wf + (((size_t)nt0 * 32 + kt) * 32 + lane) * 16;
            v16h b0 = *(const v16h*)(bp);
            v16h b1 = *(const v16h*)(bp + 16384);   // +1 N-tile  (32*32*16 fp16)
            v16h b2 = *(const v16h*)(bp + 32768);
            v16h b3 = *(const v16h*)(bp + 49152);
            acc0 = __builtin_amdgcn_wmma_f32_16x16x32_f16(false, a, false, b0, (short)0, acc0, false, false);
            acc1 = __builtin_amdgcn_wmma_f32_16x16x32_f16(false, a, false, b1, (short)0, acc1, false, false);
            acc2 = __builtin_amdgcn_wmma_f32_16x16x32_f16(false, a, false, b2, (short)0, acc2, false, false);
            acc3 = __builtin_amdgcn_wmma_f32_16x16x32_f16(false, a, false, b3, (short)0, acc3, false, false);
        }
        const int nl = lane & 15;
        const int mb = (lane >> 4) * 8;
#pragma unroll
        for (int q = 0; q < 4; ++q) {
            v8f acc = (q == 0) ? acc0 : (q == 1) ? acc1 : (q == 2) ? acc2 : acc3;
            const int g = (nt0 + q) * 16 + nl;
            const float bs = bias[g];
#pragma unroll
            for (int v = 0; v < 8; ++v) {
                int m = mt * 16 + mb + v;
                int l = m & (LDIM - 1), bI = m >> 11;
                xp[((size_t)l * BDIM + bI) * GDIM + g] = acc[v] + bs;
            }
        }
    }
}

// ---------------- persistent recurrent scan ----------------
__global__ __launch_bounds__(256, 1) void k_scan(const _Float16* __restrict__ whf,
                                                 const float* __restrict__ xp,
                                                 float* __restrict__ out,
                                                 _Float16* __restrict__ hbuf,
                                                 int* __restrict__ bar) {
    __shared__ _Float16 sW[4 * 32 * 32 * 16];   // 128 KB: this WG's W_hh slice, fragment order
    __shared__ float sG[2][16][64];             // partial gate sums (two K-halves)
    __shared__ float sC[8][16];                 // cell state

    const int tid  = threadIdx.x;
    const int lane = tid & 31, wv = tid >> 5;
    const int w    = blockIdx.x;                // 0..63, owns channels [16w, 16w+16)

    { // stage weights into LDS once via async (ASYNCcnt-tracked) copies
        unsigned long long gsrc = (unsigned long long)(uintptr_t)(whf + (size_t)w * 65536);
        unsigned ldst = (unsigned)(uintptr_t)&sW[0];   // low 32 bits = LDS byte offset
        for (int i = tid; i < 8192; i += 256) {
            unsigned long long ga = gsrc + (unsigned long long)i * 16u;
            unsigned la = ldst + (unsigned)i * 16u;
            asm volatile("global_load_async_to_lds_b128 %0, %1, off"
                         :: "v"(la), "v"(ga) : "memory");
        }
        asm volatile("s_wait_asynccnt 0x0" ::: "memory");
    }
    if (tid < 128) sC[tid >> 4][tid & 15] = 0.f;
    __syncthreads();

    const int nt = wv & 3, kh = wv >> 2;        // N-tile (gate type), K-half
    const int kb = (lane >> 4) * 8;
    const int mrow = lane & 15;
    const int chT = tid & 15, bT = tid >> 4;

    for (int t = 0; t < LDIM; ++t) {
        const _Float16* hin = hbuf + (size_t)(t & 1) * 16 * CDIM;
        _Float16* hout      = hbuf + (size_t)((t + 1) & 1) * 16 * CDIM;
        const _Float16* hrow = hin + mrow * CDIM + kb;

        // prefetch this step's x-projection gates (independent of h; hides L2 latency
        // behind the WMMA loop)
        float xg0 = 0.f, xg1 = 0.f, xg2 = 0.f, xg3 = 0.f;
        if (tid < 128) {
            const float* xpb = xp + ((size_t)t * BDIM + bT) * GDIM + w * 16 + chT;
            xg0 = xpb[0]; xg1 = xpb[1024]; xg2 = xpb[2048]; xg3 = xpb[3072];
        }

        v8f acc = {};
#pragma unroll 4
        for (int kt = 0; kt < 16; ++kt) {
            int K = (kh * 16 + kt) * 32;
            v8h alo = *(const v8h*)(hrow + K);
            v8h ahi = *(const v8h*)(hrow + K + 16);
            v16h a  = combine16(alo, ahi);
            v16h b  = *(const v16h*)(sW + ((size_t)(nt * 32 + kh * 16 + kt) * 32 + lane) * 16);
            acc = __builtin_amdgcn_wmma_f32_16x16x32_f16(false, a, false, b,
                                                         (short)0, acc, false, false);
        }
        {
            const int nl = nt * 16 + (lane & 15);
            const int mb = (lane >> 4) * 8;
#pragma unroll
            for (int v = 0; v < 8; ++v) sG[kh][mb + v][nl] = acc[v];
        }
        __syncthreads();

        if (tid < 128) {
            float gi = sG[0][bT][chT]      + sG[1][bT][chT]      + xg0;
            float gf = sG[0][bT][16 + chT] + sG[1][bT][16 + chT] + xg1;
            float gg = sG[0][bT][32 + chT] + sG[1][bT][32 + chT] + xg2;
            float go = sG[0][bT][48 + chT] + sG[1][bT][48 + chT] + xg3;
            float ii = 1.f / (1.f + __expf(-gi));
            float ff = 1.f / (1.f + __expf(-gf));
            float gv = tanhf(gg);
            float oo = 1.f / (1.f + __expf(-go));
            float c  = ff * sC[bT][chT] + ii * gv;
            sC[bT][chT] = c;
            float h  = oo * tanhf(c);
            out[((size_t)bT * LDIM + t) * CDIM + w * 16 + chT] = h;
            hout[bT * CDIM + w * 16 + chT] = (_Float16)h;
            __threadfence();
        }
        __syncthreads();

        // device-wide barrier: fresh counter per step (re-zeroed every launch)
        if (tid == 0) {
            __hip_atomic_fetch_add(&bar[t], 1, __ATOMIC_RELEASE, __HIP_MEMORY_SCOPE_AGENT);
            while (__hip_atomic_load(&bar[t], __ATOMIC_ACQUIRE, __HIP_MEMORY_SCOPE_AGENT) < NWG)
                __builtin_amdgcn_s_sleep(1);
        }
        __syncthreads();
    }
}

extern "C" void kernel_launch(void* const* d_in, const int* in_sizes, int n_in,
                              void* d_out, int out_size, void* d_ws, size_t ws_size,
                              hipStream_t stream) {
    const float* x    = (const float*)d_in[0];
    const float* Wih  = (const float*)d_in[1];
    const float* Whh  = (const float*)d_in[2];
    const float* bih  = (const float*)d_in[3];
    const float* bhh  = (const float*)d_in[4];
    float* out        = (float*)d_out;

    char* ws = (char*)d_ws;
    _Float16* x16  = (_Float16*)(ws + WS_X16);
    _Float16* wihF = (_Float16*)(ws + WS_WIH);
    _Float16* whhF = (_Float16*)(ws + WS_WHH);
    float*    bias = (float*)(ws + WS_BIAS);
    _Float16* hbuf = (_Float16*)(ws + WS_HBUF);
    int*      bar  = (int*)(ws + WS_BAR);
    float*    xprj = (float*)(ws + WS_XPROJ);

    const int NX = BDIM * LDIM * CDIM;      // 16,777,216
    const int NW = GDIM * CDIM;             // 4,194,304

    k_cvt_x   <<<(NX + 255) / 256, 256, 0, stream>>>(x, x16, NX);
    k_pack_wih<<<(NW + 255) / 256, 256, 0, stream>>>(Wih, wihF);
    k_pack_whh<<<(NW + 255) / 256, 256, 0, stream>>>(Whh, whhF);
    k_misc    <<<(2 * 16 * CDIM + 255) / 256, 256, 0, stream>>>(bih, bhh, bias, hbuf, bar);

    k_xproj<<<1024, 256, 0, stream>>>(x16, wihF, bias, xprj);
    k_scan <<<NWG, 256, 0, stream>>>(whhF, xprj, out, hbuf, bar);
}
